// MyNet_50861002719666
// MI455X (gfx1250) — matrix operand order repaired and checked by reading, not compiled, not run on hardware
//
#include <hip/hip_runtime.h>
#include <hip/hip_bf16.h>

// ---------------------------------------------------------------------------
// Continuous convolution (Open3D-style) on point clouds, CDNA5 / gfx1250.
//   N=10000 points, CIN=64, COUT=32, K=32 neighbors, radius 0.07, 3x3x3 cells.
// Pipeline:
//   K0: W f32 -> f16 (workspace)
//   K1: brute-force KNN (top-32 by d^2, radius-masked), idx + 1/nn
//       __launch_bounds__(256,1): keep the 64-reg (d2,id) ladder in VGPRs,
//       no scratch spills in the 10^8-iteration hot loop.
//   K2: per 16-query tile:
//       - TDM tensor_load_to_lds stages W (108 KB f16) into LDS, overlapped
//         with the Phase-A trilinear scatter building A[16][1728] f32 in LDS
//       - 8 waves run v_wmma_f32_16x16x32_f16 over K-slices, ds_add_f32
//         reduction, scale by 1/nn + bias.
// ---------------------------------------------------------------------------

#define NPTS    10000
#define CIN     64
#define COUT    32
#define KNB     32
#define KSIZE   3
#define NCELLS  27              // 3*3*3
#define KDIM    (NCELLS * CIN)  // 1728
#define NKCHUNK (KDIM / 32)     // 54
#define TILE_M  16
#define RADIUS  0.07f

#define WELEMS  (KDIM * COUT)   // 55296 f16 elements
#define SA_BYTES   (TILE_M * KDIM * 4)   // 110592
#define SW_BYTES   (WELEMS * 2)          // 110592
#define SOUT_BYTES (TILE_M * COUT * 4)   // 2048
#define SMEM_BYTES (SA_BYTES + SW_BYTES + SOUT_BYTES)  // 223232 < 320K

typedef __attribute__((ext_vector_type(16))) _Float16 v16h;
typedef __attribute__((ext_vector_type(8)))  float    v8f;
typedef unsigned int u32x4 __attribute__((ext_vector_type(4)));
typedef int          i32x4 __attribute__((ext_vector_type(4)));
typedef int          i32x8 __attribute__((ext_vector_type(8)));

// ---------------------------------------------------------------------------
// Kernel 0: convert W (27*64*32 f32, row-major [cell][cin][cout]) to f16.
// Flattened, W_h[(cell*64+f)*32 + o] == W[cell][f][o] -> the (K=1728)x(N=32)
// row-major B matrix for the GEMM.
// ---------------------------------------------------------------------------
__global__ void wconv_kernel(const float* __restrict__ W,
                             _Float16* __restrict__ Wh) {
    int i = blockIdx.x * blockDim.x + threadIdx.x;
    if (i < WELEMS) Wh[i] = (_Float16)W[i];
}

// ---------------------------------------------------------------------------
// Kernel 1: brute-force top-32 nearest neighbors per query.
// One thread per query; candidates staged through LDS in 256-point tiles,
// next tile prefetched (global_prefetch_b8). 32-entry (d2,id) ladder in VGPRs
// (static indexing only; launch_bounds gives the register budget for it).
// ---------------------------------------------------------------------------
__global__ void __launch_bounds__(256, 1)
knn_kernel(const float* __restrict__ points,
           int* __restrict__ idx,
           float* __restrict__ inv_nn) {
    __shared__ float sx[256], sy[256], sz[256];

    const int n = blockIdx.x * 256 + threadIdx.x;
    const bool active = (n < NPTS);

    // inactive lanes scan with garbage query point; results discarded
    const int pn = active ? n : 0;
    const float px = points[pn * 3 + 0];
    const float py = points[pn * 3 + 1];
    const float pz = points[pn * 3 + 2];

    float bd[KNB];
    int   bi[KNB];
#pragma unroll
    for (int k = 0; k < KNB; ++k) { bd[k] = 3.4e38f; bi[k] = -1; }

    for (int base = 0; base < NPTS; base += 256) {
        __syncthreads();
        const int j = base + threadIdx.x;
        if (j < NPTS) {
            sx[threadIdx.x] = points[j * 3 + 0];
            sy[threadIdx.x] = points[j * 3 + 1];
            sz[threadIdx.x] = points[j * 3 + 2];
        }
        // prefetch next candidate tile (L2-resident working set)
        const int jn = j + 256;
        if (jn < NPTS) __builtin_prefetch(points + jn * 3, 0, 1);
        __syncthreads();

        const int cnt = (NPTS - base < 256) ? (NPTS - base) : 256;
        for (int t = 0; t < cnt; ++t) {
            const float dx = sx[t] - px;
            const float dy = sy[t] - py;
            const float dz = sz[t] - pz;
            const float d2 = dx * dx + dy * dy + dz * dz;
            if (d2 < bd[KNB - 1]) {
                bd[KNB - 1] = d2;
                bi[KNB - 1] = base + t;
                // single bubble pass restores sortedness (array was sorted)
#pragma unroll
                for (int s = KNB - 1; s > 0; --s) {
                    if (bd[s] < bd[s - 1]) {
                        float td = bd[s]; bd[s] = bd[s - 1]; bd[s - 1] = td;
                        int   ti = bi[s]; bi[s] = bi[s - 1]; bi[s - 1] = ti;
                    }
                }
            }
        }
    }

    if (active) {
        const float R2 = RADIUS * RADIUS;
        int nn = 0;
#pragma unroll
        for (int k = 0; k < KNB; ++k) {
            const bool v = (bi[k] >= 0) && (bd[k] <= R2);
            idx[n * KNB + k] = v ? bi[k] : -1;
            nn += v ? 1 : 0;
        }
        inv_nn[n] = 1.0f / fmaxf((float)nn, 1.0f);
    }
}

// ---------------------------------------------------------------------------
// TDM: stage the full W_h (55296 f16 = 110592 B) into LDS at lds_off.
// D# : 2D tensor, data_size=2B, one row of 55296 elements (tile_dim0 fits
// the 16-bit field), type=2 ("image"). Issued by one wave; EXEC ignored.
// ---------------------------------------------------------------------------
__device__ __forceinline__ void tdm_load_w_to_lds(const _Float16* Wh,
                                                  unsigned lds_off) {
    const unsigned long long ga = (unsigned long long)(uintptr_t)Wh;
    u32x4 g0;
    g0[0] = 1u;                                   // count=1, user descriptor
    g0[1] = lds_off;                              // lds_addr (bytes)
    g0[2] = (unsigned)(ga & 0xffffffffu);         // global_addr[31:0]
    g0[3] = (unsigned)((ga >> 32) & 0x01ffffffu)  // global_addr[56:32]
          | (2u << 30);                           // type=2
    i32x8 g1;
    g1[0] = 0x00010000;                 // wg_mask=0, data_size=1 (2 bytes)
    g1[1] = (int)((unsigned)WELEMS << 16);  // tensor_dim0[15:0] at bit 48
    g1[2] = 0x00010000;                 // tensor_dim0[31:16]=0, tensor_dim1=1
    g1[3] = (int)((unsigned)WELEMS << 16);  // tile_dim0 = 55296
    g1[4] = 1;                          // tile_dim1 = 1, tile_dim2 = 0
    g1[5] = (int)WELEMS;                // tensor_dim0_stride[31:0]
    g1[6] = 0;
    g1[7] = 0;
    const i32x4 gz = {0, 0, 0, 0};
#if defined(__clang_major__) && (__clang_major__ >= 23)
    const i32x8 gz8 = {0, 0, 0, 0, 0, 0, 0, 0};
    __builtin_amdgcn_tensor_load_to_lds(g0, g1, gz, gz, gz8, 0);
#else
    __builtin_amdgcn_tensor_load_to_lds(g0, g1, gz, gz, 0);
#endif
}

// ---------------------------------------------------------------------------
// Kernel 2: per-tile scatter + WMMA GEMM.
// Block = 512 threads = 16 wave32; one block handles 16 queries.
// Phase A: wave w owns query w's A row (1728 f32 in LDS); lanes own 2
//          features each -> plain LDS read-modify-write, no atomics.
//          Meanwhile the TDM copies W_h into LDS (async, TENSORcnt).
// Phase B: waves 0..7 -> (col tile = wave&1, K-slice = wave>>1 of 4);
//          ~14 x v_wmma_f32_16x16x32_f16 each, ds_add_f32 reduction.
// ---------------------------------------------------------------------------
extern __shared__ char smem_raw[];   // SMEM_BYTES

__global__ void __launch_bounds__(512, 1)
cconv_kernel(const float* __restrict__ feats,
             const float* __restrict__ points,
             const _Float16* __restrict__ Wh,
             const float* __restrict__ bias,
             const int* __restrict__ idx,
             const float* __restrict__ inv_nn,
             float* __restrict__ out) {
    float*    s_A   = (float*)smem_raw;                      // [16][1728]
    _Float16* s_W   = (_Float16*)(smem_raw + SA_BYTES);      // [1728][32]
    float*    s_out = (float*)(smem_raw + SA_BYTES + SW_BYTES); // [16][32]

    const int wave = threadIdx.x >> 5;   // 0..15
    const int lane = threadIdx.x & 31;   // 0..31
    const int tile = blockIdx.x;         // 0..624 (10000/16 exact)

    // ---------------- init + async W staging ----------------
    s_out[threadIdx.x & (TILE_M * COUT - 1)] = 0.0f;   // 512 threads, 512 slots

    if (wave == 0) {
        // dynamic LDS starts at offset 0 (no static LDS in this kernel)
        tdm_load_w_to_lds(Wh, (unsigned)SA_BYTES);
    }

    // ---------------- Phase A: build A rows ----------------
    const int q  = wave;                     // query within tile
    const int gq = tile * TILE_M + q;        // global query id
    float* Arow  = s_A + q * KDIM;

    for (int i = lane; i < KDIM; i += 32) Arow[i] = 0.0f;

    const float px = points[gq * 3 + 0];
    const float py = points[gq * 3 + 1];
    const float pz = points[gq * 3 + 2];
    const float invR = 1.0f / RADIUS;

    for (int k = 0; k < KNB; ++k) {
        const int id = idx[gq * KNB + k];
        if (id < 0) continue;

        // scalar geometry, computed redundantly per lane (cheap)
        const float rx = (points[id * 3 + 0] - px) * invR;
        const float ry = (points[id * 3 + 1] - py) * invR;
        const float rz = (points[id * 3 + 2] - pz) * invR;
        const float l2   = sqrtf(rx * rx + ry * ry + rz * rz);
        const float linf = fmaxf(fabsf(rx), fmaxf(fabsf(ry), fabsf(rz)));
        const float s    = (linf > 0.0f) ? (l2 / fmaxf(linf, 1e-12f)) : 0.0f;
        // coords = (t+1)*0.5*(KSIZE-1) = t+1 for KSIZE=3
        const float cx = rx * s + 1.0f;
        const float cy = ry * s + 1.0f;
        const float cz = rz * s + 1.0f;
        const float c0x = fminf(fmaxf(floorf(cx), 0.0f), (float)(KSIZE - 2));
        const float c0y = fminf(fmaxf(floorf(cy), 0.0f), (float)(KSIZE - 2));
        const float c0z = fminf(fmaxf(floorf(cz), 0.0f), (float)(KSIZE - 2));
        const float fx = cx - c0x, fy = cy - c0y, fz = cz - c0z;
        const int ix = (int)c0x, iy = (int)c0y, iz = (int)c0z;

        // each lane owns features lane and lane+32
        const float f0 = feats[id * CIN + lane];
        const float f1 = feats[id * CIN + lane + 32];

#pragma unroll
        for (int c = 0; c < 8; ++c) {
            const int oi = (c >> 2) & 1, oj = (c >> 1) & 1, ok = c & 1;
            const float w = (oi ? fx : 1.0f - fx) *
                            (oj ? fy : 1.0f - fy) *
                            (ok ? fz : 1.0f - fz);
            const int cell = (ix + oi) * 9 + (iy + oj) * 3 + (iz + ok);
            float* p = Arow + cell * CIN;
            p[lane]      += w * f0;
            p[lane + 32] += w * f1;
        }
    }

    // W staging must be complete before Phase B reads s_W
    if (wave == 0) __builtin_amdgcn_s_wait_tensorcnt(0);
    __syncthreads();

    // ---------------- Phase B: WMMA GEMM ----------------
    if (wave < 8) {
        const int ncb    = (wave & 1) * 16;  // output column base
        const int kslice = wave >> 1;        // 0..3, K-chunk stride 4
        const int m      = lane & 15;
        // A-fragment as loaded here (contiguous k order, same as B): lanes 0-15 hold K 0..15,
        // lanes 16-31 hold K 16..31 of row M = lane&15.
        const int kbaseA = (lane < 16) ? 0 : 16;
        // B-fragment as loaded here (contiguous k order, same as A): lanes 0-15 hold K 0..15 of col N=lane,
        // lanes 16-31 hold K 16..31 of col N=lane-16.
        const int kbaseB = (lane < 16) ? 0 : 16;
        const float* Am = s_A + m * KDIM;

        v8f acc = {};
        for (int kc = kslice; kc < NKCHUNK; kc += 4) {
            const int kk = kc * 32;
            v16h a, b;
#pragma unroll
            for (int h = 0; h < 16; ++h) {
                const int ko = kbaseA + h;
                a[h] = (_Float16)Am[kk + ko];
            }
#pragma unroll
            for (int h = 0; h < 16; ++h) {
                b[h] = s_W[(kk + kbaseB + h) * COUT + ncb + m];
            }
            acc = __builtin_amdgcn_wmma_f32_16x16x32_f16(
                false, a, false, b, (short)0, acc, false, false);
        }

        // D layout: VGPR r, lane l -> M = r + (l<16 ? 0 : 8), N = l&15
        // reduce K-slice partials across waves via LDS float atomics
#pragma unroll
        for (int r = 0; r < 8; ++r) {
            const int M = r + ((lane < 16) ? 0 : 8);
            atomicAdd(&s_out[M * COUT + ncb + m], acc[r]);
        }
    }

    __syncthreads();

    // ---------------- epilogue: scale + bias ----------------
    {
        const int t   = threadIdx.x;          // 512 == 16*32 exact
        const int M   = t >> 5;
        const int col = t & 31;
        const int g   = tile * TILE_M + M;
        out[g * COUT + col] = s_out[t] * inv_nn[g] + bias[col];
    }
}

// ---------------------------------------------------------------------------
// Launch
// ---------------------------------------------------------------------------
extern "C" void kernel_launch(void* const* d_in, const int* in_sizes, int n_in,
                              void* d_out, int out_size, void* d_ws, size_t ws_size,
                              hipStream_t stream) {
    const float* feats  = (const float*)d_in[0];   // N*CIN
    const float* points = (const float*)d_in[1];   // N*3
    const float* W      = (const float*)d_in[2];   // 27*CIN*COUT
    const float* bias   = (const float*)d_in[3];   // COUT
    float* out = (float*)d_out;                    // N*COUT

    // workspace layout (16B aligned):
    //   [0)          idx    : N*KNB int   = 1,280,000 B
    //   [1,280,000)  inv_nn : N float     =    40,000 B
    //   [1,320,000)  W_h    : 55296 f16   =   110,592 B
    char* ws = (char*)d_ws;
    int*      w_idx   = (int*)(ws);
    float*    w_invnn = (float*)(ws + 1280000);
    _Float16* w_Wh    = (_Float16*)(ws + 1320000);

    // K0: convert weights to f16
    wconv_kernel<<<(WELEMS + 255) / 256, 256, 0, stream>>>(W, w_Wh);

    // K1: KNN
    knn_kernel<<<(NPTS + 255) / 256, 256, 0, stream>>>(points, w_idx, w_invnn);

    // K2: scatter + WMMA GEMM (dynamic LDS: A + W_h + out tile)
    const int tiles = NPTS / TILE_M;               // 625, exact
    cconv_kernel<<<tiles, TILE_M * 32, SMEM_BYTES, stream>>>(
        feats, points, w_Wh, bias, w_idx, w_invnn, out);
}
